// MutiHeadAttention_5952824673153
// MI455X (gfx1250) — matrix operand order repaired and checked by reading, not compiled, not run on hardware
//
#include <hip/hip_runtime.h>
#include <hip/hip_bf16.h>
#include <hip/hip_fp16.h>

// Problem constants (from reference)
#define B_  2
#define S_  2048
#define D_  1024
#define H_  16
#define DH_ 64
#define M_  (B_*S_)                          // 4096 rows for projection GEMMs
#define HEAD_ELEMS ((size_t)B_*H_*S_*DH_)    // 4 Mi f16 elements per Q/K/V tensor
#define WMAT_ELEMS ((size_t)D_*D_)           // 1 Mi f16 elements per weight

typedef __attribute__((ext_vector_type(16))) _Float16 v16h;
typedef __attribute__((ext_vector_type(8)))  float    v8f;
typedef __attribute__((ext_vector_type(4)))  unsigned int v4u;
typedef __attribute__((ext_vector_type(8)))  int      v8i;
typedef __attribute__((ext_vector_type(4)))  int      v4i;

// ---- CDNA5 async data-movement feature probes ------------------------------
#if __has_builtin(__builtin_amdgcn_global_load_async_to_lds_b128)
#define HAVE_ASYNC_LDS 1
#endif
#if __has_builtin(__builtin_amdgcn_tensor_load_to_lds)
#define HAVE_TDM 1
#endif

#if __has_builtin(__builtin_amdgcn_s_wait_asynccnt)
#define WAIT_ASYNC() __builtin_amdgcn_s_wait_asynccnt(0)
#else
#define WAIT_ASYNC() asm volatile("s_wait_asynccnt 0x0" ::: "memory")
#endif
#if __has_builtin(__builtin_amdgcn_s_wait_tensorcnt)
#define WAIT_TENSOR() __builtin_amdgcn_s_wait_tensorcnt(0)
#else
#define WAIT_TENSOR() asm volatile("s_wait_tensorcnt 0x0" ::: "memory")
#endif

#ifdef HAVE_ASYNC_LDS
typedef __attribute__((address_space(1))) v4i* gptr_v4i;   // global (prints as __device__)
typedef __attribute__((address_space(3))) v4i* lptr_v4i;   // LDS    (prints as __shared__)
__device__ __forceinline__ void async_cp16(const void* g, void* l) {
  // one 16-byte global -> LDS async copy per lane (GLOBAL_LOAD_ASYNC_TO_LDS_B128)
  __builtin_amdgcn_global_load_async_to_lds_b128((gptr_v4i)g, (lptr_v4i)l, 0, 0);
}
#endif

__device__ __forceinline__ v8f wmma16(v16h a, v16h b, v8f c) {
  // D = A(16x32 f16) * B(32x16 f16) + C(16x16 f32)
  return __builtin_amdgcn_wmma_f32_16x16x32_f16(false, a, false, b, (short)0, c, false, false);
}

// ---------------------------------------------------------------------------
// Kernel 0: convert the four weight matrices f32 [K][N] -> f16 transposed [N][K].
// ---------------------------------------------------------------------------
__global__ void __launch_bounds__(256)
wcvt_kernel(const float* __restrict__ Wq, const float* __restrict__ Wk,
            const float* __restrict__ Wv, const float* __restrict__ Wo,
            _Float16* __restrict__ wt) {
  const int which = blockIdx.z;
  const float* W = (which == 0) ? Wq : (which == 1) ? Wk : (which == 2) ? Wv : Wo;
  _Float16* dst = wt + (size_t)which * WMAT_ELEMS;

  __shared__ _Float16 tile[16][17];          // +1 pad to dodge bank conflicts
  const int tx = threadIdx.x & 15;
  const int ty = threadIdx.x >> 4;
  const int k0 = blockIdx.x * 16;
  const int n0 = blockIdx.y * 16;

  tile[ty][tx] = (_Float16)W[(size_t)(k0 + ty) * D_ + (n0 + tx)];
  __syncthreads();
  dst[(size_t)(n0 + ty) * D_ + (k0 + tx)] = tile[tx][ty];
}

// ---------------------------------------------------------------------------
// Kernel 1: QKV projections.  Y = X * W  (X: [4096,1024] f32, Wt: f16 [N][K]).
// One wave computes a 16x64 tile (4 WMMA accumulators, A reused 4x).
// Q,K stored f16 per-head [B][H][S][DH]; V stored transposed [B][H][DH][S].
// ---------------------------------------------------------------------------
__global__ void __launch_bounds__(32)
qkv_proj_kernel(const float* __restrict__ Xq, const float* __restrict__ Xk,
                const float* __restrict__ Xv,
                const _Float16* __restrict__ wt,
                _Float16* __restrict__ ws) {
  const int which = blockIdx.z;
  const float* X = (which == 0) ? Xq : ((which == 1) ? Xk : Xv);
  const _Float16* Wt = wt + (size_t)which * WMAT_ELEMS;   // [N][K]

  const int r0 = blockIdx.x * 16;
  const int j0 = blockIdx.y * 64;          // one head per column tile
  const int lane = threadIdx.x & 31;
  const int mn = lane & 15;
  const int g  = lane >> 4;

  v8f acc[4] = {{}, {}, {}, {}};
  for (int k0 = 0; k0 < D_; k0 += 32) {
    v16h a;
    const float* xrow = X + (size_t)(r0 + mn) * D_ + k0;
    __builtin_prefetch(xrow + 32, 0, 1);
    #pragma unroll
    for (int e = 0; e < 8; ++e) {
      a[e]     = (_Float16)xrow[16*g + e];
      a[8 + e] = (_Float16)xrow[16*g + 8 + e];
    }
    #pragma unroll
    for (int nt = 0; nt < 4; ++nt) {
      const _Float16* wrow = Wt + (size_t)(j0 + 16*nt + mn) * D_ + k0 + 16*g;
      if (nt == 0) __builtin_prefetch(wrow + 32, 0, 1);
      v16h bt;
      #pragma unroll
      for (int e = 0; e < 16; ++e) bt[e] = wrow[e];
      acc[nt] = wmma16(a, bt, acc[nt]);
    }
  }

  const int h  = j0 >> 6;
  const int b  = r0 >> 11;
  const int s0 = r0 & 2047;
  if (which < 2) {                         // Q, K: [B][H][S][DH]
    _Float16* dst = ws + (size_t)which * HEAD_ELEMS;
    #pragma unroll
    for (int nt = 0; nt < 4; ++nt)
      #pragma unroll
      for (int i = 0; i < 8; ++i) {
        const int s = s0 + i + 8*g;
        dst[(((size_t)b * H_ + h) * S_ + s) * DH_ + 16*nt + mn] = (_Float16)acc[nt][i];
      }
  } else {                                 // V transposed: [B][H][DH][S]
    _Float16* dstT = ws + 2 * HEAD_ELEMS;
    #pragma unroll
    for (int nt = 0; nt < 4; ++nt)
      #pragma unroll
      for (int i = 0; i < 8; ++i) {
        const int s = s0 + i + 8*g;
        dstT[(((size_t)b * H_ + h) * DH_ + 16*nt + mn) * S_ + s] = (_Float16)acc[nt][i];
      }
  }
}

// ---------------------------------------------------------------------------
// Kernel 2: flash-style masked attention.
// Block = 4 waves = 4 query tiles of one (b,h).  K chunk DMA'd to LDS by the
// Tensor Data Mover (wave 0), Vt chunk copied with async global->LDS loads;
// all 4 waves consume the shared LDS chunks.  Online softmax per wave.
// ---------------------------------------------------------------------------
__global__ void __launch_bounds__(128)
attention_kernel(const _Float16* __restrict__ ws_qkv,
                 _Float16* __restrict__ attn_out,
                 const int* __restrict__ valid_lens) {
  const int h  = blockIdx.y;
  const int b  = blockIdx.z;
  const int tid  = threadIdx.x;
  const int w    = tid >> 5;
  const int lane = tid & 31;
  const int mn = lane & 15;
  const int g  = lane >> 4;

  const size_t headoff = (((size_t)b * H_) + h) * (size_t)S_ * DH_;
  const _Float16* Q  = ws_qkv + headoff;                    // [S][64]
  const _Float16* K  = ws_qkv + HEAD_ELEMS + headoff;       // [S][64]
  const _Float16* Vt = ws_qkv + 2 * HEAD_ELEMS + headoff;   // [64][S]
  const int vlen = valid_lens[b];
  const int q0 = (blockIdx.x * 4 + w) * 16;

  __shared__ _Float16 kbuf[32][64];         // K chunk  (4 KB, contiguous in memory)
  __shared__ _Float16 vbuf[64][32];         // Vt chunk (4 KB)
  __shared__ _Float16 pbuf[4][16][32];      // per-wave P staging (4 KB)

  // Resident Q A-tiles: aq0 d=0..31, aq1 d=32..63
  v16h aq0, aq1;
  {
    const _Float16* qrow = Q + (size_t)(q0 + mn) * DH_;
    #pragma unroll
    for (int e = 0; e < 8; ++e) {
      aq0[e]     = qrow[16*g + e];
      aq0[8 + e] = qrow[16*g + 8 + e];
      aq1[e]     = qrow[32 + 16*g + e];
      aq1[8 + e] = qrow[40 + 16*g + e];
    }
  }

  v8f acc0 = {}, acc1 = {}, acc2 = {}, acc3 = {};
  float mrow[8], lrow[8];
  #pragma unroll
  for (int i = 0; i < 8; ++i) { mrow[i] = -3.0e38f; lrow[i] = 0.0f; }

  for (int kv0 = 0; kv0 < S_; kv0 += 32) {
    __syncthreads();                        // previous chunk fully consumed

    // ---- K chunk: one TDM descriptor moves the contiguous 4 KB tile --------
#ifdef HAVE_TDM
    if (w == 0) {
      const unsigned lds_addr = (unsigned)(size_t)&kbuf[0][0];
      const unsigned long long ga =
          (unsigned long long)(size_t)(K + (size_t)kv0 * DH_);
      v4u g0;
      g0[0] = 1u;                                        // count=1 (user mode)
      g0[1] = lds_addr;                                  // lds_addr [63:32]
      g0[2] = (unsigned)(ga & 0xffffffffu);              // global_addr lo
      g0[3] = (unsigned)((ga >> 32) & 0x1ffffffu)        // global_addr [56:32]
              | (2u << 30);                              // type=2 ("image")
      const unsigned td0 = 2048, td1 = 64;               // tensor dims (elements)
      const unsigned tile0 = 2048, tile1 = 1;            // tile: one 2048-elem row
      const unsigned long long st0 = 2048;               // dim0 stride
      v8i g1;
      g1[0] = (int)(1u << 16);                           // data_size=1 (2 bytes)
      g1[1] = (int)((td0 & 0xffffu) << 16);              // tensor_dim0[15:0]
      g1[2] = (int)((td0 >> 16) | ((td1 & 0xffffu) << 16));
      g1[3] = (int)((td1 >> 16) | (tile0 << 16));        // tile_dim0
      g1[4] = (int)tile1;                                // tile_dim1 (tile_dim2=0)
      g1[5] = (int)(st0 & 0xffffffffu);                  // stride0 lo
      g1[6] = (int)((st0 >> 32) & 0xffffu);              // stride0 hi | stride1 lo
      g1[7] = 0;
      v4i g2 = 0, g3 = 0;
#if __clang_major__ >= 23
      v8i g4 = 0;
      __builtin_amdgcn_tensor_load_to_lds(g0, g1, g2, g3, g4, 0);
#else
      __builtin_amdgcn_tensor_load_to_lds(g0, g1, g2, g3, 0);
#endif
    }
#else
    {
      const uint4* ksrc = (const uint4*)(K + (size_t)kv0 * DH_);
      uint4* kdst = (uint4*)&kbuf[0][0];
      for (int t = tid; t < 256; t += 128) kdst[t] = ksrc[t];
    }
#endif

    // ---- Vt chunk: async global->LDS copies (ASYNCcnt) ---------------------
#ifdef HAVE_ASYNC_LDS
    for (int t = tid; t < 256; t += 128) {
      const int row = t >> 2, seg = t & 3;
      async_cp16(Vt + (size_t)row * S_ + kv0 + seg * 8, &vbuf[0][0] + t * 8);
    }
    WAIT_ASYNC();
#else
    {
      uint4* vdst = (uint4*)&vbuf[0][0];
      for (int t = tid; t < 256; t += 128) {
        const int row = t >> 2, seg = t & 3;
        vdst[t] = *(const uint4*)(Vt + (size_t)row * S_ + kv0 + seg * 8);
      }
    }
#endif
#ifdef HAVE_TDM
    if (w == 0) WAIT_TENSOR();
#endif
    __syncthreads();                        // LDS chunks visible to all waves

    // ---- scores: two 16-col tiles, K-dim = 64 -> 2 WMMA each (B from LDS) ----
    v8f sc0 = {}, sc1 = {};
    {
      v16h bk0, bk1;
      #pragma unroll
      for (int e = 0; e < 16; ++e) { bk0[e] = kbuf[mn][16*g + e];
                                     bk1[e] = kbuf[mn][32 + 16*g + e]; }
      sc0 = wmma16(aq0, bk0, sc0);
      sc0 = wmma16(aq1, bk1, sc0);
      #pragma unroll
      for (int e = 0; e < 16; ++e) { bk0[e] = kbuf[16 + mn][16*g + e];
                                     bk1[e] = kbuf[16 + mn][32 + 16*g + e]; }
      sc1 = wmma16(aq0, bk0, sc1);
      sc1 = wmma16(aq1, bk1, sc1);
    }

    // ---- scale (1/sqrt(64)) + valid_lens mask ----
    const bool ok0 = (kv0 + mn)      < vlen;
    const bool ok1 = (kv0 + 16 + mn) < vlen;
    #pragma unroll
    for (int i = 0; i < 8; ++i) {
      sc0[i] = ok0 ? sc0[i] * 0.125f : -1.0e6f;
      sc1[i] = ok1 ? sc1[i] * 0.125f : -1.0e6f;
    }

    // ---- chunk row-max (butterfly within each 16-lane half) ----
    float cm[8];
    #pragma unroll
    for (int i = 0; i < 8; ++i) cm[i] = fmaxf(sc0[i], sc1[i]);
    #pragma unroll
    for (int off = 1; off < 16; off <<= 1) {
      #pragma unroll
      for (int i = 0; i < 8; ++i) cm[i] = fmaxf(cm[i], __shfl_xor(cm[i], off, 32));
    }

    // ---- online softmax update ----
    float p0[8], p1[8], rs[8];
    #pragma unroll
    for (int i = 0; i < 8; ++i) {
      const float mnew  = fmaxf(mrow[i], cm[i]);
      const float scale = __expf(mrow[i] - mnew);
      mrow[i] = mnew;
      lrow[i] *= scale;
      p0[i] = __expf(sc0[i] - mnew);
      p1[i] = __expf(sc1[i] - mnew);
      rs[i] = p0[i] + p1[i];
      acc0[i] *= scale; acc1[i] *= scale; acc2[i] *= scale; acc3[i] *= scale;
    }
    #pragma unroll
    for (int off = 1; off < 16; off <<= 1) {
      #pragma unroll
      for (int i = 0; i < 8; ++i) rs[i] += __shfl_xor(rs[i], off, 32);
    }
    #pragma unroll
    for (int i = 0; i < 8; ++i) lrow[i] += rs[i];

    // ---- P: C-layout -> per-wave LDS -> A-layout (16x32 f16) ----
    #pragma unroll
    for (int i = 0; i < 8; ++i) {
      pbuf[w][i + 8*g][mn]      = (_Float16)p0[i];
      pbuf[w][i + 8*g][16 + mn] = (_Float16)p1[i];
    }
    __syncthreads();
    v16h ap;
    #pragma unroll
    for (int e = 0; e < 8; ++e) {
      ap[e]     = pbuf[w][mn][16*g + e];
      ap[8 + e] = pbuf[w][mn][16*g + 8 + e];
    }

    // ---- P(16x32) x V(32x64): 4 WMMAs, B tiles contiguous from LDS Vt ----
    #pragma unroll
    for (int dt = 0; dt < 4; ++dt) {
      v16h bv;
      #pragma unroll
      for (int e = 0; e < 16; ++e) bv[e] = vbuf[dt*16 + mn][16*g + e];
      if (dt == 0) acc0 = wmma16(ap, bv, acc0);
      else if (dt == 1) acc1 = wmma16(ap, bv, acc1);
      else if (dt == 2) acc2 = wmma16(ap, bv, acc2);
      else acc3 = wmma16(ap, bv, acc3);
    }
  }

  // ---- epilogue: normalize, store concat-heads f16 [B][S][D] ----
  #pragma unroll
  for (int i = 0; i < 8; ++i) {
    const float inv = 1.0f / lrow[i];
    const int s = q0 + i + 8*g;
    const size_t base = ((size_t)b * S_ + s) * D_ + (size_t)h * DH_ + mn;
    attn_out[base + 0]  = (_Float16)(acc0[i] * inv);
    attn_out[base + 16] = (_Float16)(acc1[i] * inv);
    attn_out[base + 32] = (_Float16)(acc2[i] * inv);
    attn_out[base + 48] = (_Float16)(acc3[i] * inv);
  }
}

// ---------------------------------------------------------------------------
// Kernel 3: output projection.  out = A(f16 [4096,1024]) * Wo^T(f16) -> f32.
// ---------------------------------------------------------------------------
__global__ void __launch_bounds__(32)
out_proj_kernel(const _Float16* __restrict__ Ain, const _Float16* __restrict__ WoT,
                float* __restrict__ out) {
  const int r0 = blockIdx.x * 16;
  const int j0 = blockIdx.y * 64;
  const int lane = threadIdx.x & 31;
  const int mn = lane & 15;
  const int g  = lane >> 4;

  v8f acc[4] = {{}, {}, {}, {}};
  for (int k0 = 0; k0 < D_; k0 += 32) {
    v16h a;
    const _Float16* arow = Ain + (size_t)(r0 + mn) * D_ + k0;
    __builtin_prefetch(arow + 32, 0, 1);
    #pragma unroll
    for (int e = 0; e < 8; ++e) {
      a[e]     = arow[16*g + e];
      a[8 + e] = arow[16*g + 8 + e];
    }
    #pragma unroll
    for (int nt = 0; nt < 4; ++nt) {
      const _Float16* wrow = WoT + (size_t)(j0 + 16*nt + mn) * D_ + k0 + 16*g;
      v16h bt;
      #pragma unroll
      for (int e = 0; e < 16; ++e) bt[e] = wrow[e];
      acc[nt] = wmma16(a, bt, acc[nt]);
    }
  }
  #pragma unroll
  for (int nt = 0; nt < 4; ++nt)
    #pragma unroll
    for (int i = 0; i < 8; ++i)
      out[(size_t)(r0 + i + 8*g) * D_ + (j0 + 16*nt + mn)] = acc[nt][i];
}

// ---------------------------------------------------------------------------
extern "C" void kernel_launch(void* const* d_in, const int* in_sizes, int n_in,
                              void* d_out, int out_size, void* d_ws, size_t ws_size,
                              hipStream_t stream) {
  const float* queries    = (const float*)d_in[0];
  const float* keys       = (const float*)d_in[1];
  const float* values     = (const float*)d_in[2];
  const float* Wq         = (const float*)d_in[3];
  const float* Wk         = (const float*)d_in[4];
  const float* Wv         = (const float*)d_in[5];
  const float* Wo         = (const float*)d_in[6];
  const int*   valid_lens = (const int*)d_in[7];

  // Workspace layout (f16 elements):
  //   [0, 3*HEAD)         Qh, Kh, Vt
  //   [3*HEAD, 4*HEAD)    attention output (heads concat, [B][S][D])
  //   [4*HEAD, +4*D*D)    4 transposed f16 weight matrices
  _Float16* ws_qkv   = (_Float16*)d_ws;
  _Float16* attn_out = ws_qkv + 3 * HEAD_ELEMS;
  _Float16* wtrans   = ws_qkv + 4 * HEAD_ELEMS;

  wcvt_kernel<<<dim3(D_/16, D_/16, 4), dim3(256), 0, stream>>>(
      Wq, Wk, Wv, Wo, wtrans);

  qkv_proj_kernel<<<dim3(M_/16, D_/64, 3), dim3(32), 0, stream>>>(
      queries, keys, values, wtrans, ws_qkv);

  attention_kernel<<<dim3(S_/64, H_, B_), dim3(128), 0, stream>>>(
      ws_qkv, attn_out, valid_lens);

  out_proj_kernel<<<dim3(M_/16, D_/64), dim3(32), 0, stream>>>(
      attn_out, wtrans + 3 * WMAT_ELEMS, (float*)d_out);
}
